// CAModule_53309134078084
// MI455X (gfx1250) — matrix operand order repaired and checked by reading, not compiled, not run on hardware
//
#include <hip/hip_runtime.h>

#define B_  4
#define C_  256
#define CH_ 128
#define N_  4096

typedef __attribute__((ext_vector_type(16))) __bf16 v16bf;
typedef __attribute__((ext_vector_type(8)))  __bf16 v8bf;
typedef __attribute__((ext_vector_type(8)))  float  v8f;
typedef __attribute__((ext_vector_type(4)))  unsigned int v4u;
typedef __attribute__((ext_vector_type(8)))  int v8i;
typedef __attribute__((ext_vector_type(4)))  int v4i;

#if defined(__has_builtin)
#  if __has_builtin(__builtin_amdgcn_tensor_load_to_lds)
#    define USE_TDM 1
#  endif
#endif
#ifndef USE_TDM
#  define USE_TDM 0
#endif

__device__ __forceinline__ v16bf combine16(v8bf lo, v8bf hi) {
  v16bf r;
#pragma unroll
  for (int i = 0; i < 8; i++) { r[i] = lo[i]; r[i + 8] = hi[i]; }
  return r;
}

__device__ __forceinline__ v8f wmma_bf16(v16bf a, v16bf b, v8f c) {
  return __builtin_amdgcn_wmma_f32_16x16x32_bf16(
      /*neg_a=*/false, a, /*neg_b=*/false, b,
      /*c_mod=*/(short)0, c, /*reuse_a=*/false, /*reuse_b=*/false);
}

#if USE_TDM
// Issue one TDM 2D tile load (global -> LDS).  Descriptor per ISA 8.3/8.4:
// group0 = {count=1, lds_addr, global_addr[56:0], type=2}
// group1 = {mask=0, data_size=1 (2B), tensor_dim0/1, tile_dim0/1, dim0_stride}
// groups 2/3 zero (2D tile, no iterate/gather/pad).
__device__ __forceinline__ void tdm_load_2d(unsigned lds_addr, const void* gptr,
                                            unsigned tensor_d0, unsigned tensor_d1,
                                            unsigned tile_d0, unsigned tile_d1,
                                            unsigned long long d0_stride) {
  const unsigned long long ga = (unsigned long long)gptr;
  v4u g0;
  g0[0] = 1u;                                             // count=1
  g0[1] = lds_addr;                                       // LDS byte address
  g0[2] = (unsigned)(ga & 0xffffffffu);                   // global_addr lo
  g0[3] = (unsigned)((ga >> 32) & 0x1ffffffu) | (2u << 30); // ga hi | type=2
  v8i g1;
  g1[0] = (int)(1u << 16);                                // data_size = 2 bytes
  g1[1] = (int)((tensor_d0 & 0xffffu) << 16);             // tensor_dim0[15:0]
  g1[2] = (int)((tensor_d0 >> 16) | ((tensor_d1 & 0xffffu) << 16));
  g1[3] = (int)((tensor_d1 >> 16) | (tile_d0 << 16));     // | tile_dim0
  g1[4] = (int)(tile_d1 & 0xffffu);                       // tile_dim1, tile_dim2=0
  g1[5] = (int)(d0_stride & 0xffffffffu);                 // dim0_stride lo32
  g1[6] = (int)((d0_stride >> 32) & 0xffffu);             // dim0_stride hi16
  g1[7] = 0;
  v4i z4 = (v4i)0;
#if __clang_major__ >= 23
  v8i z8 = (v8i)0;
  __builtin_amdgcn_tensor_load_to_lds(g0, g1, z4, z4, z8, 0);
#else
  __builtin_amdgcn_tensor_load_to_lds(g0, g1, z4, z4, 0);
#endif
}
#endif  // USE_TDM

// ---------------------------------------------------------------------------
// Kernel 1: QKV projections. One wave computes one 16n x 16ch output tile of
// one of {Q,K,V}.  Q,K stored [b][n][c] (row-major in c); V stored [b][c][n]
// so the attention kernel's B-operands are contiguous 16-byte loads.
// ---------------------------------------------------------------------------
__global__ __launch_bounds__(128) void qkv_proj_kernel(
    const float* __restrict__ xquery, const float* __restrict__ xkey,
    const float* __restrict__ wq, const float* __restrict__ bq,
    const float* __restrict__ wk, const float* __restrict__ bk,
    const float* __restrict__ wv, const float* __restrict__ bv,
    __bf16* __restrict__ Qt, __bf16* __restrict__ Kt, __bf16* __restrict__ Vt) {
  const int wave = threadIdx.x >> 5;
  const int lane = threadIdx.x & 31;
  const int l16  = lane & 15;
  const int half = lane >> 4;

  int gid = blockIdx.x * 4 + wave;
  const int mat = gid % 3; gid /= 3;          // 0=Q, 1=K, 2=V
  const int cht = gid % (CH_ / 16); gid /= (CH_ / 16);
  const int nt  = gid % (N_ / 16);
  const int b   = gid / (N_ / 16);

  const float* x    = (mat == 0) ? xquery : xkey;
  const float* w    = (mat == 0) ? wq : (mat == 1 ? wk : wv);
  const float* bias = (mat == 0) ? bq : (mat == 1 ? bk : bv);

  const int n0 = nt * 16, ch0 = cht * 16;
  const float bval = bias[ch0 + l16];
  v8f acc;
#pragma unroll
  for (int i = 0; i < 8; i++) acc[i] = bval;   // bias per output column

  const float* xb = x + (size_t)b * C_ * N_;
  const int n_row = n0 + l16;
#pragma unroll
  for (int cc = 0; cc < C_ / 32; cc++) {
    const int c0 = cc * 32;
    v16bf a;
#pragma unroll
    for (int e = 0; e < 8; e++)
      a[e] = (__bf16)xb[(size_t)(c0 + 16 * half + e) * N_ + n_row];
#pragma unroll
    for (int e = 0; e < 8; e++)
      a[8 + e] = (__bf16)xb[(size_t)(c0 + 16 * half + 8 + e) * N_ + n_row];
    const float* wrow = w + (size_t)(ch0 + l16) * C_ + c0 + 16 * half;
    v16bf bm;
#pragma unroll
    for (int j = 0; j < 16; j++) bm[j] = (__bf16)wrow[j];
    acc = wmma_bf16(a, bm, acc);
  }

  if (mat < 2) {
    __bf16* dst = (mat == 0) ? Qt : Kt;
#pragma unroll
    for (int i = 0; i < 8; i++) {
      const int n = n0 + i + 8 * half;
      dst[((size_t)b * N_ + n) * CH_ + ch0 + l16] = (__bf16)acc[i];
    }
  } else {
#pragma unroll
    for (int i = 0; i < 8; i++) {
      const int n = n0 + i + 8 * half;
      Vt[((size_t)b * CH_ + ch0 + l16) * N_ + n] = (__bf16)acc[i];
    }
  }
}

// ---------------------------------------------------------------------------
// Kernel 2: flash attention + output projection + residual, fused.
// Block = 4 waves; wave w owns 16 queries.  K/V tiles (32 keys) are staged in
// double-buffered LDS by the Tensor Data Mover (wave 0 issues, TENSORcnt +
// barrier publishes), so the WG fetches each tile exactly once while the
// previous tile's 16 WMMAs + online softmax execute.
// ---------------------------------------------------------------------------
__global__ __launch_bounds__(128) void flash_attn_out_kernel(
    const __bf16* __restrict__ Qt, const __bf16* __restrict__ Kt,
    const __bf16* __restrict__ Vt,
    const float* __restrict__ wo, const float* __restrict__ bo,
    const float* __restrict__ xquery, float* __restrict__ out) {
  __shared__ __align__(16) __bf16 lds_k[2][32][CH_];   // [buf][k][c]  8KB each
  __shared__ __align__(16) __bf16 lds_v[2][CH_][32];   // [buf][c][k]  8KB each
  __shared__ __align__(16) __bf16 lds_p[4][16][CH_];   // per-wave P/O transpose

  const int wave = threadIdx.x >> 5;
  const int lane = threadIdx.x & 31;
  const int l16  = lane & 15;
  const int half = lane >> 4;

  const int b  = blockIdx.x / (N_ / 64);
  const int qb = blockIdx.x % (N_ / 64);
  const int q0 = qb * 64 + wave * 16;

  // stage a 32-key K/V tile pair into LDS buffer `buf`
  auto fill = [&](int buf, int k0) {
#if USE_TDM
    if (wave == 0) {
      tdm_load_2d((unsigned)(size_t)&lds_k[buf][0][0],
                  Kt + ((size_t)b * N_ + k0) * CH_,
                  /*tensor_d0=*/CH_, /*tensor_d1=*/N_,
                  /*tile_d0=*/CH_, /*tile_d1=*/32, /*d0_stride=*/CH_);
      tdm_load_2d((unsigned)(size_t)&lds_v[buf][0][0],
                  Vt + (size_t)b * CH_ * N_ + k0,
                  /*tensor_d0=*/N_, /*tensor_d1=*/CH_,
                  /*tile_d0=*/32, /*tile_d1=*/CH_, /*d0_stride=*/N_);
    }
#else
    const int tid = threadIdx.x;
    // K tile is one contiguous 8KB block ([n][c] layout, 32 consecutive rows)
    const v8bf* gk = (const v8bf*)(Kt + ((size_t)b * N_ + k0) * CH_);
    v8bf* sk = (v8bf*)&lds_k[buf][0][0];
#pragma unroll
    for (int i = 0; i < 4; i++) sk[tid + 128 * i] = gk[tid + 128 * i];
    // V tile: 128 rows of 64B, row stride N_
    {
      const v8bf* gv = (const v8bf*)(Vt + ((size_t)b * CH_ + tid) * N_ + k0);
      v8bf* sv = (v8bf*)&lds_v[buf][tid][0];
#pragma unroll
      for (int i = 0; i < 4; i++) sv[i] = gv[i];
    }
#endif
  };

  // Q tile in A-layout registers (4 chunks of K=32 over c=128)
  v16bf Aq[4];
  const __bf16* qrow = Qt + ((size_t)b * N_ + q0 + l16) * CH_;
#pragma unroll
  for (int ch = 0; ch < 4; ch++) {
    const v8bf* p0 = (const v8bf*)(qrow + ch * 32 + 16 * half);
    const v8bf* p1 = (const v8bf*)(qrow + ch * 32 + 16 * half + 8);
    Aq[ch] = combine16(*p0, *p1);
  }

  v8f o_acc[8];
#pragma unroll
  for (int t = 0; t < 8; t++)
#pragma unroll
    for (int i = 0; i < 8; i++) o_acc[t][i] = 0.f;
  float row_m[8], row_l[8];
#pragma unroll
  for (int i = 0; i < 8; i++) { row_m[i] = -3.0e38f; row_l[i] = 0.f; }

  fill(0, 0);
  for (int kb = 0; kb < N_ / 32; kb++) {
    const int cur = kb & 1;
#if USE_TDM
    __builtin_amdgcn_s_wait_tensorcnt(0);   // buf[cur] DMA complete (wave 0)
#endif
    __syncthreads();                        // publish buf[cur]; reads of
                                            // buf[1-cur] from kb-1 are done
    if (kb + 1 < N_ / 32) fill(1 - cur, (kb + 1) * 32);

    v8f s0, s1;
#pragma unroll
    for (int i = 0; i < 8; i++) { s0[i] = 0.f; s1[i] = 0.f; }

    // S = Q^T K : B-operand column k = LDS row (contiguous in c)
    const __bf16* k0row = &lds_k[cur][l16][0];
    const __bf16* k1row = &lds_k[cur][16 + l16][0];
#pragma unroll
    for (int ch = 0; ch < 4; ch++) {
      const v8bf* p0 = (const v8bf*)(k0row + ch * 32 + 16 * half);
      const v8bf* p1 = (const v8bf*)(k0row + ch * 32 + 16 * half + 8);
      s0 = wmma_bf16(Aq[ch], combine16(*p0, *p1), s0);
    }
#pragma unroll
    for (int ch = 0; ch < 4; ch++) {
      const v8bf* p0 = (const v8bf*)(k1row + ch * 32 + 16 * half);
      const v8bf* p1 = (const v8bf*)(k1row + ch * 32 + 16 * half + 8);
      s1 = wmma_bf16(Aq[ch], combine16(*p0, *p1), s1);
    }

    // ---- online softmax over these 32 columns (rows live in half-waves) ----
    float mnew[8], alpha[8], psum[8], p0v[8], p1v[8];
#pragma unroll
    for (int i = 0; i < 8; i++) mnew[i] = fmaxf(s0[i], s1[i]);
#pragma unroll
    for (int st = 1; st < 16; st <<= 1)
#pragma unroll
      for (int i = 0; i < 8; i++)
        mnew[i] = fmaxf(mnew[i], __shfl_xor(mnew[i], st, 32));
#pragma unroll
    for (int i = 0; i < 8; i++) {
      mnew[i]  = fmaxf(mnew[i], row_m[i]);
      alpha[i] = __expf(row_m[i] - mnew[i]);
      row_m[i] = mnew[i];
      p0v[i]   = __expf(s0[i] - mnew[i]);
      p1v[i]   = __expf(s1[i] - mnew[i]);
      psum[i]  = p0v[i] + p1v[i];
    }
#pragma unroll
    for (int st = 1; st < 16; st <<= 1)
#pragma unroll
      for (int i = 0; i < 8; i++) psum[i] += __shfl_xor(psum[i], st, 32);
#pragma unroll
    for (int i = 0; i < 8; i++) row_l[i] = row_l[i] * alpha[i] + psum[i];
#pragma unroll
    for (int t = 0; t < 8; t++)
#pragma unroll
      for (int i = 0; i < 8; i++) o_acc[t][i] *= alpha[i];

    // P tile: D-layout -> LDS [q][k] -> A-layout (wave-local; DS in-order)
#pragma unroll
    for (int i = 0; i < 8; i++) {
      lds_p[wave][i + 8 * half][l16]      = (__bf16)p0v[i];
      lds_p[wave][i + 8 * half][16 + l16] = (__bf16)p1v[i];
    }
    asm volatile("s_wait_dscnt 0" ::: "memory");
    v16bf Pa;
    {
      const v8bf* pp0 = (const v8bf*)(&lds_p[wave][l16][16 * half]);
      const v8bf* pp1 = (const v8bf*)(&lds_p[wave][l16][16 * half + 8]);
      Pa = combine16(*pp0, *pp1);
    }

    // O += P * V^T : B-operand column c = LDS-V row (contiguous in k)
#pragma unroll
    for (int t = 0; t < 8; t++) {
      const __bf16* vrow = &lds_v[cur][t * 16 + l16][16 * half];
      const v8bf* p0 = (const v8bf*)(vrow);
      const v8bf* p1 = (const v8bf*)(vrow + 8);
      o_acc[t] = wmma_bf16(Pa, combine16(*p0, *p1), o_acc[t]);
    }
  }

  // normalize by softmax denominator
#pragma unroll
  for (int i = 0; i < 8; i++) {
    const float inv = 1.0f / row_l[i];
#pragma unroll
    for (int t = 0; t < 8; t++) o_acc[t][i] *= inv;
  }

  // stash O (16q x 128c) in LDS as bf16 for the A-operand of the out-proj
#pragma unroll
  for (int t = 0; t < 8; t++)
#pragma unroll
    for (int i = 0; i < 8; i++)
      lds_p[wave][i + 8 * half][t * 16 + l16] = (__bf16)o_acc[t][i];
  asm volatile("s_wait_dscnt 0" ::: "memory");

  // final = wo @ O + bo + x_query  (16 o-tiles x 4 c-chunks = 64 WMMAs)
  const float* xrow = xquery + (size_t)b * C_ * N_;
  float*       orow = out    + (size_t)b * C_ * N_;
  for (int ot = 0; ot < C_ / 16; ot++) {
    const int o = ot * 16 + l16;
    const float bv = bo[o];
    v8f acc;
#pragma unroll
    for (int i = 0; i < 8; i++) acc[i] = bv;
#pragma unroll
    for (int ch = 0; ch < 4; ch++) {
      const v8bf* a0 = (const v8bf*)(&lds_p[wave][l16][ch * 32 + 16 * half]);
      const v8bf* a1 = (const v8bf*)(&lds_p[wave][l16][ch * 32 + 16 * half + 8]);
      const float* wrow = wo + (size_t)o * CH_ + ch * 32 + 16 * half;
      v16bf bw;
#pragma unroll
      for (int j = 0; j < 16; j++) bw[j] = (__bf16)wrow[j];
      acc = wmma_bf16(combine16(*a0, *a1), bw, acc);
    }
#pragma unroll
    for (int i = 0; i < 8; i++) {
      const int q = q0 + i + 8 * half;
      const size_t idx = (size_t)o * N_ + q;
      orow[idx] = acc[i] + xrow[idx];
    }
  }
}

// ---------------------------------------------------------------------------
extern "C" void kernel_launch(void* const* d_in, const int* in_sizes, int n_in,
                              void* d_out, int out_size, void* d_ws,
                              size_t ws_size, hipStream_t stream) {
  (void)in_sizes; (void)n_in; (void)out_size; (void)ws_size;
  const float* x_query = (const float*)d_in[0];
  const float* x_key   = (const float*)d_in[1];
  const float* wq = (const float*)d_in[2];
  const float* bq = (const float*)d_in[3];
  const float* wk = (const float*)d_in[4];
  const float* bk = (const float*)d_in[5];
  const float* wv = (const float*)d_in[6];
  const float* bv = (const float*)d_in[7];
  const float* wo = (const float*)d_in[8];
  const float* bo = (const float*)d_in[9];
  float* out = (float*)d_out;

  const size_t mat_bytes = (size_t)B_ * N_ * CH_ * 2;   // 4 MiB each
  __bf16* Qt = (__bf16*)d_ws;
  __bf16* Kt = (__bf16*)((char*)d_ws + mat_bytes);
  __bf16* Vt = (__bf16*)((char*)d_ws + 2 * mat_bytes);

  const int waves1 = B_ * (N_ / 16) * (CH_ / 16) * 3;
  qkv_proj_kernel<<<waves1 / 4, 128, 0, stream>>>(
      x_query, x_key, wq, bq, wk, bk, wv, bv, Qt, Kt, Vt);

  flash_attn_out_kernel<<<B_ * (N_ / 64), 128, 0, stream>>>(
      Qt, Kt, Vt, wo, bo, x_query, out);
}
